// Attention_72705206387306
// MI455X (gfx1250) — matrix operand order repaired and checked by reading, not compiled, not run on hardware
//
#include <hip/hip_runtime.h>

typedef __attribute__((ext_vector_type(16))) __bf16        v16bf;
typedef __attribute__((ext_vector_type(8)))  float         v8f;
typedef __attribute__((ext_vector_type(8)))  unsigned int  v8u;
typedef __attribute__((ext_vector_type(4)))  unsigned int  v4u;

#define WAVES 8
#define BQ    128      // q rows per workgroup (8 waves x 16)
#define TK    32       // kv columns per step
#define NSEQ  4096
#define DHEAD 128
#define NT    (NSEQ / TK)

// padded LDS row lengths (elements); strides stay 16B-aligned but avoid
// 64-bank aliasing: 136*2B=272B (bank step 4), 40*2B=80B (bank step 20)
#define KS_LD 136
#define VS_LD 40
#define PS_LD 40

// fp32 -> bf16, round-half-up
__device__ __forceinline__ unsigned short f2bf(float f) {
    unsigned int u = __builtin_bit_cast(unsigned int, f);
    return (unsigned short)((u + 0x8000u) >> 16);
}
// pack two fp32 -> packed bf16x2 with one v_perm_b32
__device__ __forceinline__ unsigned int pack2bf(float lo, float hi) {
    unsigned int a = __builtin_bit_cast(unsigned int, lo) + 0x8000u;
    unsigned int b = __builtin_bit_cast(unsigned int, hi) + 0x8000u;
    return __builtin_amdgcn_perm(b, a, 0x07060302u);  // {b[31:16], a[31:16]}
}

// 16 contiguous bf16 (32 bytes) from LDS -> fragment
__device__ __forceinline__ v16bf ld_frag32(const unsigned short* p) {
    v8u u = *(const v8u*)p;
    return __builtin_bit_cast(v16bf, u);
}
// two 16-byte halves (A-fragment split: K and K+16) -> fragment
__device__ __forceinline__ v16bf ld_frag_split(const unsigned short* p0,
                                               const unsigned short* p1) {
    v4u lo = *(const v4u*)p0;
    v4u hi = *(const v4u*)p1;
    v8u u;
    u[0]=lo[0]; u[1]=lo[1]; u[2]=lo[2]; u[3]=lo[3];
    u[4]=hi[0]; u[5]=hi[1]; u[6]=hi[2]; u[7]=hi[3];
    return __builtin_bit_cast(v16bf, u);
}

// DPP16 row_ror:N (rotate within each 16-lane row -> per-half-wave reduce)
template <int CTRL>
__device__ __forceinline__ float dpp_mov(float x) {
    int xi = __builtin_bit_cast(int, x);
    return __builtin_bit_cast(float,
        __builtin_amdgcn_update_dpp(xi, xi, CTRL, 0xf, 0xf, true));
}
__device__ __forceinline__ float rowmax16(float x) {
    x = fmaxf(x, dpp_mov<0x128>(x));
    x = fmaxf(x, dpp_mov<0x124>(x));
    x = fmaxf(x, dpp_mov<0x122>(x));
    x = fmaxf(x, dpp_mov<0x121>(x));
    return x;
}
__device__ __forceinline__ float rowsum16(float x) {
    x += dpp_mov<0x128>(x);
    x += dpp_mov<0x124>(x);
    x += dpp_mov<0x122>(x);
    x += dpp_mov<0x121>(x);
    return x;
}

// pack 16 staged floats into 16 bf16 and store as 2x b128
__device__ __forceinline__ void store16bf(unsigned short* dst, const float* a) {
    unsigned int w[8];
    #pragma unroll
    for (int i = 0; i < 8; ++i) w[i] = pack2bf(a[2*i], a[2*i+1]);
    *(v4u*)dst       = (v4u){w[0], w[1], w[2], w[3]};
    *(v4u*)(dst + 8) = (v4u){w[4], w[5], w[6], w[7]};
}

__global__ __launch_bounds__(256, 1)
void attn_fa_bf16_kernel(const float* __restrict__ Q, const float* __restrict__ K,
                         const float* __restrict__ V, float* __restrict__ O)
{
    __shared__ unsigned short Ks[2][TK][KS_LD];     // K tiles, row-major [kvcol][d]
    __shared__ unsigned short Vs[2][DHEAD][VS_LD];  // V tiles, transposed [d][kvcol]
    __shared__ unsigned short Ps[WAVES][16][PS_LD]; // per-wave P scratch

    const int b    = blockIdx.y;
    const int qblk = blockIdx.x;
    const int tid  = (int)threadIdx.x;
    const int wave = tid >> 5;
    const int lane = tid & 31;
    const int hl   = lane >> 4;
    const int l16  = lane & 15;

    // softmax(s/sqrt(d)) = exp2(c*(s-m)), c = log2(e)/sqrt(128)
    const float c = 1.4426950408889634f * 0.08838834764831845f;

    const float* Qg = Q + ((size_t)b * NSEQ + (size_t)qblk * BQ + wave * 16) * DHEAD;
    const float* Kg = K + (size_t)b * NSEQ * DHEAD;
    const float* Vg = V + (size_t)b * NSEQ * DHEAD;

    // ---- Q fragments (A-layout, bf16), loaded once, kept in VGPRs ----
    v16bf qf[4];
    {
        const float* qrow = Qg + (size_t)l16 * DHEAD;
        #pragma unroll
        for (int s = 0; s < 4; ++s) {
            float qv[16];
            #pragma unroll
            for (int j = 0; j < 8; ++j) qv[j]     = qrow[s*32 +      hl*8 + j];
            #pragma unroll
            for (int j = 0; j < 8; ++j) qv[8 + j] = qrow[s*32 + 16 + hl*8 + j];
            v8u u;
            #pragma unroll
            for (int i = 0; i < 8; ++i) u[i] = pack2bf(qv[2*i], qv[2*i+1]);
            qf[s] = __builtin_bit_cast(v16bf, u);
        }
    }

    // ---- running state ----
    v8f o[8];
    #pragma unroll
    for (int d = 0; d < 8; ++d) o[d] = (v8f){0.f,0.f,0.f,0.f,0.f,0.f,0.f,0.f};
    float mrow[8], lrow[8];
    #pragma unroll
    for (int v = 0; v < 8; ++v) { mrow[v] = -3.0e38f; lrow[v] = 0.0f; }

    // staging maps
    const int krow0 = tid >> 3;          // K: kv row (0..31)
    const int kcol0 = (tid & 7) * 16;    // K: 16 head-dim cols
    const int vcol  = tid & 127;         // V: head-dim column (0..127)
    const int vk0   = (tid >> 7) * 16;   // V: kv range base (0 or 16)

    // ---- prologue: stage tile 0 into buffer 0 ----
    {
        float ka[16], va[16];
        const float4* kn = (const float4*)(Kg + (size_t)krow0 * DHEAD + kcol0);
        #pragma unroll
        for (int i = 0; i < 4; ++i) ((float4*)ka)[i] = kn[i];
        #pragma unroll
        for (int j = 0; j < 16; ++j) va[j] = Vg[(size_t)(vk0 + j) * DHEAD + vcol];
        store16bf(&Ks[0][krow0][kcol0], ka);
        store16bf(&Vs[0][vcol][vk0],    va);
    }

    for (int kb = 0; kb < NT; ++kb) {
        __syncthreads();                  // buf[kb&1] ready; buf[(kb+1)&1] free
        const int p = kb & 1;

        // ---- issue next tile's global loads early (land during compute) ----
        float ka[16], va[16];
        const bool have_next = (kb + 1 < NT);
        if (have_next) {
            const float4* kn = (const float4*)(Kg + ((size_t)(kb+1)*TK + krow0)*DHEAD + kcol0);
            #pragma unroll
            for (int i = 0; i < 4; ++i) ((float4*)ka)[i] = kn[i];
            const float* vbase = Vg + ((size_t)(kb+1)*TK + vk0)*DHEAD + vcol;
            #pragma unroll
            for (int j = 0; j < 16; ++j) va[j] = vbase[(size_t)j * DHEAD];
            if (kb + 2 < NT) {            // -> global_prefetch_b8 for tile kb+2
                __builtin_prefetch((const float*)kn + (size_t)TK*DHEAD, 0, 0);
                __builtin_prefetch(vbase + (size_t)TK*DHEAD, 0, 0);
            }
        }

        // ---- S(16x32) = Qtile x Ktile^T : 8 WMMAs ----
        v16bf kf0[4], kf1[4];
        #pragma unroll
        for (int s = 0; s < 4; ++s) {
            kf0[s] = ld_frag_split(&Ks[p][l16     ][s*32 + hl*8], &Ks[p][l16     ][s*32 + 16 + hl*8]);
            kf1[s] = ld_frag_split(&Ks[p][16 + l16][s*32 + hl*8], &Ks[p][16 + l16][s*32 + 16 + hl*8]);
        }
        v8f acc0 = (v8f){0.f,0.f,0.f,0.f,0.f,0.f,0.f,0.f};
        v8f acc1 = (v8f){0.f,0.f,0.f,0.f,0.f,0.f,0.f,0.f};
        #pragma unroll
        for (int s = 0; s < 4; ++s) {
            acc0 = __builtin_amdgcn_wmma_f32_16x16x32_bf16(false, qf[s], false, kf0[s],
                                                           (short)0, acc0, false, false);
            acc1 = __builtin_amdgcn_wmma_f32_16x16x32_bf16(false, qf[s], false, kf1[s],
                                                           (short)0, acc1, false, false);
        }

        // ---- online softmax (DPP row reductions) ----
        #pragma unroll
        for (int v = 0; v < 8; ++v) {
            float s0 = acc0[v], s1 = acc1[v];
            float mx    = rowmax16(fmaxf(s0, s1));
            float mnew  = fmaxf(mrow[v], mx);
            float alpha = __builtin_amdgcn_exp2f((mrow[v] - mnew) * c);
            float p0    = __builtin_amdgcn_exp2f((s0 - mnew) * c);
            float p1    = __builtin_amdgcn_exp2f((s1 - mnew) * c);
            float rs    = rowsum16(p0 + p1);
            lrow[v] = lrow[v] * alpha + rs;
            mrow[v] = mnew;
            #pragma unroll
            for (int d = 0; d < 8; ++d) o[d][v] *= alpha;
            const int pr = v + 8 * hl;
            Ps[wave][pr][l16]      = f2bf(p0);
            Ps[wave][pr][16 + l16] = f2bf(p1);
        }
        // same-wave LDS RAW: DS pipeline is in-order; fence compiler + counter
        __builtin_amdgcn_wave_barrier();
        asm volatile("s_wait_dscnt 0" ::: "memory");

        // ---- O(16x128) += P(16x32) x Vtile(32x128) : 8 WMMAs ----
        v16bf pf = ld_frag_split(&Ps[wave][l16][hl*8], &Ps[wave][l16][16 + hl*8]);
        v16bf vbf[8];
        #pragma unroll
        for (int d = 0; d < 8; ++d)
            vbf[d] = ld_frag_split(&Vs[p][d*16 + l16][hl*8], &Vs[p][d*16 + l16][16 + hl*8]);
        #pragma unroll
        for (int d = 0; d < 8; ++d)
            o[d] = __builtin_amdgcn_wmma_f32_16x16x32_bf16(false, pf, false, vbf[d],
                                                           (short)0, o[d], false, false);

        // ---- convert + store next tile into the other buffer ----
        if (have_next) {
            store16bf(&Ks[p^1][krow0][kcol0], ka);
            store16bf(&Vs[p^1][vcol][vk0],    va);
        }
    }

    // ---- epilogue: normalize by row sums and store fp32 ----
    float* Og = O + ((size_t)b * NSEQ + (size_t)qblk * BQ + wave * 16) * DHEAD;
    #pragma unroll
    for (int v = 0; v < 8; ++v) {
        const int r = v + 8 * hl;
        const float inv = 1.0f / lrow[v];
        #pragma unroll
        for (int d = 0; d < 8; ++d)
            Og[(size_t)r * DHEAD + d*16 + l16] = o[d][v] * inv;
    }
}

extern "C" void kernel_launch(void* const* d_in, const int* in_sizes, int n_in,
                              void* d_out, int out_size, void* d_ws, size_t ws_size,
                              hipStream_t stream) {
    const float* Q = (const float*)d_in[0];
    const float* K = (const float*)d_in[1];
    const float* V = (const float*)d_in[2];
    float* out = (float*)d_out;
    const int B = in_sizes[0] / (NSEQ * DHEAD);   // 16
    dim3 grid(NSEQ / BQ, B);                      // (32, 16)
    attn_fa_bf16_kernel<<<grid, dim3(256), 0, stream>>>(Q, K, V, out);
}
